// MultiHeadAttention_15281493639244
// MI455X (gfx1250) — matrix operand order repaired and checked by reading, not compiled, not run on hardware
//
#include <hip/hip_runtime.h>
#include <hip/hip_bf16.h>

// ---------------------------------------------------------------------------
// MHA forward for MI455X (gfx1250, wave32, WMMA bf16 16x16x32, f32 accum)
// B=8, S=1024, D=1024, H=16, depth=64
// ---------------------------------------------------------------------------

typedef __attribute__((ext_vector_type(16))) __bf16 v16bf;
typedef __attribute__((ext_vector_type(8)))  float  v8f;

#define WMMA_BF16(a, b, c) \
  __builtin_amdgcn_wmma_f32_16x16x32_bf16(false, (a), false, (b), (short)0, (c), false, false)

constexpr int Bc = 8;
constexpr int Sc = 1024;
constexpr int Dc = 1024;
constexpr int Hc = 16;
constexpr int DEPTH = 64;

// ----- CDNA5 async global->LDS (guarded; manual copy fallback) -------------
#if __has_builtin(__builtin_amdgcn_global_load_async_to_lds_b128)
#define HAS_ASYNC_LDS 1
#endif

#define AS1P __attribute__((address_space(1)))
#define AS3P __attribute__((address_space(3)))

typedef int vsi4 __attribute__((__vector_size__(16)));  // 16-byte int vector

__device__ __forceinline__ void async_copy16(const void* g, void* l) {
#if defined(HAS_ASYNC_LDS)
  __builtin_amdgcn_global_load_async_to_lds_b128(
      (AS1P vsi4*)(unsigned long long)g,
      (AS3P vsi4*)(unsigned)(unsigned long long)l, 0, 0);
#else
  (void)g; (void)l;
#endif
}

__device__ __forceinline__ void async_wait0() {
#if defined(HAS_ASYNC_LDS)
#if __has_builtin(__builtin_amdgcn_s_wait_asynccnt)
  __builtin_amdgcn_s_wait_asynccnt(0);
#else
  asm volatile("s_wait_asynccnt 0" ::: "memory");
#endif
#endif
}

__device__ __forceinline__ unsigned pack_bf16(float x0, float x1) {
  union { __bf16 h[2]; unsigned u; } p;
  p.h[0] = (__bf16)x0;
  p.h[1] = (__bf16)x1;
  return p.u;
}

// ---------------------------------------------------------------------------
// WMMA fragment loaders (wave32; A and B both use contiguous k = 16*(lane>>4) + i)
// ---------------------------------------------------------------------------

// A-matrix 16x32 bf16 from row-major LDS tile (ld elements per row).
__device__ __forceinline__ v16bf frag_a_bf16(const __bf16* base, int ld, int lane) {
  const int m = lane & 15, g = lane >> 4;
  const __bf16* p = base + m * ld;
  v16bf f;
#pragma unroll
  for (int i = 0; i < 8; ++i) f[i] = p[g * 16 + i];
#pragma unroll
  for (int i = 0; i < 8; ++i) f[i + 8] = p[g * 16 + 8 + i];
  return f;
}

// B-matrix 32x16 bf16 from LDS stored transposed as [n][k].
__device__ __forceinline__ v16bf frag_b_bf16(const __bf16* baseT, int ld, int lane) {
  const int n = lane & 15, g = lane >> 4;
  const __bf16* p = baseT + n * ld + g * 16;
  v16bf f;
#pragma unroll
  for (int i = 0; i < 16; ++i) f[i] = p[i];
  return f;
}

// ---------------------------------------------------------------------------
// GEMM: Out = X(MxK) * W(KxN) + bias.  M=8192, N=K=1024.
// Block: 256 threads (8 waves, 4x2), tile 128x128, BK=32; 32x64 per wave.
// ---------------------------------------------------------------------------
template <bool IN_BF16, bool OUT_HS>
__global__ __launch_bounds__(256) void wmma_gemm_kernel(
    const void* __restrict__ Xv, const float* __restrict__ W,
    const float* __restrict__ bias, void* __restrict__ Outv) {
  constexpr int N = Dc, K = Dc;
  constexpr int BK = 32, LDA = BK + 8, LDB = BK + 8;  // 80B rows, 16B aligned

  __shared__ __bf16 As[128 * LDA];  // X tile, row-major [128][BK]
  __shared__ __bf16 BT[128 * LDB];  // W tile, transposed [n][k]

  const int tid = threadIdx.x;
  const int lane = tid & 31, wid = tid >> 5;
  const int wm = wid >> 1, wn = wid & 1;  // 4x2 waves: 32 rows x 64 cols each
  const int m0 = blockIdx.y * 128;
  const int n0 = blockIdx.x * 128;

  const float* Xf = (const float*)Xv;
  const __bf16* Xb = (const __bf16*)Xv;

  v8f acc[2][4] = {};

  for (int kt = 0; kt < K / BK; ++kt) {
    const int k0 = kt * BK;
    __syncthreads();

    // Prefetch next k-tile while staging this one (global_prefetch_b8).
    if (kt + 1 < K / BK) {
      const int k1 = k0 + BK;
      const int ar = tid >> 1, ah = (tid & 1) * 16;
      if (IN_BF16)
        __builtin_prefetch(Xb + (size_t)(m0 + ar) * K + k1 + ah, 0, 0);
      else
        __builtin_prefetch(Xf + (size_t)(m0 + ar) * K + k1 + ah, 0, 0);
      __builtin_prefetch(W + (size_t)(k1 + (tid >> 3)) * N + n0 + (tid & 7) * 16, 0, 0);
    }

    // Stage A: 128x32 as 2048 bf16-pairs (8 pairs/thread, dword LDS stores)
#pragma unroll
    for (int i = 0; i < 8; ++i) {
      int e = tid + i * 256;
      int r = e >> 4, c2 = (e & 15) * 2;
      unsigned pk;
      if (IN_BF16) {
        pk = *(const unsigned*)(Xb + (size_t)(m0 + r) * K + k0 + c2);
      } else {
        float2 x = *(const float2*)(Xf + (size_t)(m0 + r) * K + k0 + c2);
        pk = pack_bf16(x.x, x.y);
      }
      *(unsigned*)&As[r * LDA + c2] = pk;
    }
    // Stage B transposed: 32x128 as 2048 k-pairs (8 pairs/thread)
#pragma unroll
    for (int i = 0; i < 8; ++i) {
      int p = tid + i * 256;
      int nc = p & 127, kk = (p >> 7) * 2;
      float w0 = W[(size_t)(k0 + kk) * N + n0 + nc];
      float w1 = W[(size_t)(k0 + kk + 1) * N + n0 + nc];
      *(unsigned*)&BT[nc * LDB + kk] = pack_bf16(w0, w1);
    }
    __syncthreads();

    v16bf a0 = frag_a_bf16(As + (wm * 32) * LDA, LDA, lane);
    v16bf a1 = frag_a_bf16(As + (wm * 32 + 16) * LDA, LDA, lane);
#pragma unroll
    for (int j = 0; j < 4; ++j) {
      v16bf bj = frag_b_bf16(BT + (wn * 64 + j * 16) * LDB, LDB, lane);
      acc[0][j] = WMMA_BF16(a0, bj, acc[0][j]);
      acc[1][j] = WMMA_BF16(a1, bj, acc[1][j]);
    }
  }

  // Store. C layout: N = lane&15 ; VGPR r -> M = r + 8*(lane>>4).
  const int g = lane >> 4;
#pragma unroll
  for (int sm = 0; sm < 2; ++sm) {
#pragma unroll
    for (int sn = 0; sn < 4; ++sn) {
      int gcol = n0 + wn * 64 + sn * 16 + (lane & 15);
      float bv = bias[gcol];
#pragma unroll
      for (int r = 0; r < 8; ++r) {
        int grow = m0 + wm * 32 + sm * 16 + r + 8 * g;
        float val = acc[sm][sn][r] + bv;
        if (OUT_HS) {
          int bb = grow >> 10, s = grow & 1023;   // S = 1024
          int h = gcol >> 6, d = gcol & 63;       // depth = 64
          ((__bf16*)Outv)[((((size_t)bb * Hc) + h) * Sc + s) * DEPTH + d] = (__bf16)val;
        } else {
          ((float*)Outv)[(size_t)grow * N + gcol] = val;
        }
      }
    }
  }
}

// ---------------------------------------------------------------------------
// Fused attention: per block = (b, h, 32 query rows).
//   scores(32x1024) = Qh*Kh^T/8 + mask  -> softmax in LDS -> attn to d_out
//   out(32x64) = attn * Vh              -> concat bf16 [B,S,D]
// LDS: Q 32x72 bf16 | KV 64x72 bf16 | scores 32x1024 f32 | reductions
//      | bf16 probs 32x1040  == 212,736 B (of 320 KB/WGP)
// ---------------------------------------------------------------------------
constexpr int LDQ = DEPTH + 8;   // 72 elems = 144 B rows (16B aligned)
constexpr int LDK = DEPTH + 8;
constexpr int LDP = Sc + 16;     // 1040 elems = 2080 B rows (16B aligned)
constexpr size_t OFF_SC  = (size_t)(32 * LDQ + 64 * LDK) * 2;           // 13824
constexpr size_t OFF_PB  = OFF_SC + (size_t)(32 * 1024 + 256 + 64) * 4; // 146176
constexpr size_t ATTN_SMEM = OFF_PB + (size_t)32 * LDP * 2;             // 212736

__global__ __launch_bounds__(256) void attention_kernel(
    const __bf16* __restrict__ Qh, const __bf16* __restrict__ Kh,
    const __bf16* __restrict__ Vh, const float* __restrict__ mask,
    float* __restrict__ attn_out, __bf16* __restrict__ concat) {
  extern __shared__ char smem_raw[];
  __bf16* qs = (__bf16*)smem_raw;                 // [32][LDQ]
  __bf16* kv = qs + 32 * LDQ;                     // [64][LDK]
  float* sc = (float*)(smem_raw + OFF_SC);        // [32][1024]
  float* red = sc + 32 * 1024;                    // [32][8]
  float* rowmax = red + 256;                      // [32]
  float* rowsum = rowmax + 32;                    // [32]
  __bf16* pbuf = (__bf16*)(smem_raw + OFF_PB);    // [32][LDP] bf16 probs

  const int tid = threadIdx.x;
  const int lane = tid & 31, wid = tid >> 5;
  const int wm = wid >> 2, wn = wid & 3;          // 2x4 wave grid
  const int qt = blockIdx.x, h = blockIdx.y, b = blockIdx.z;
  const int q0 = qt * 32;

  const __bf16* Qb = Qh + ((((size_t)b * Hc) + h) * Sc + q0) * DEPTH;
  const __bf16* Kb = Kh + (((size_t)b * Hc) + h) * Sc * DEPTH;
  const __bf16* Vb = Vh + (((size_t)b * Hc) + h) * Sc * DEPTH;

  // Load 32x64 Q tile: 256 x 16B (1 uint4 per thread)
  {
    int r = tid >> 3, c4 = tid & 7;
#if defined(HAS_ASYNC_LDS)
    async_copy16(Qb + (size_t)r * DEPTH + c4 * 8, qs + r * LDQ + c4 * 8);
#else
    *(uint4*)(qs + r * LDQ + c4 * 8) = *(const uint4*)(Qb + (size_t)r * DEPTH + c4 * 8);
#endif
  }

  // ---- Pass 1: scores = Q K^T * (1/sqrt(64)) + mask * -1e9 ----
  for (int ch = 0; ch < 16; ++ch) {
    __syncthreads();
    // Stage K chunk 64x64: 512 x 16B (2 uint4 per thread), async if available
#pragma unroll
    for (int i = 0; i < 2; ++i) {
      int u = tid + i * 256;
      int r = u >> 3, c4 = u & 7;
#if defined(HAS_ASYNC_LDS)
      async_copy16(Kb + (size_t)(ch * 64 + r) * DEPTH + c4 * 8, kv + r * LDK + c4 * 8);
#else
      *(uint4*)(kv + r * LDK + c4 * 8) =
          *(const uint4*)(Kb + (size_t)(ch * 64 + r) * DEPTH + c4 * 8);
#endif
    }
    if (ch + 1 < 16)
      __builtin_prefetch(Kb + (size_t)(ch * 64 + 64 + (tid >> 2)) * DEPTH + (tid & 3) * 16, 0, 0);
    async_wait0();
    __syncthreads();

    v8f acc = {};
#pragma unroll
    for (int ks = 0; ks < 2; ++ks) {
      v16bf a = frag_a_bf16(qs + (wm * 16) * LDQ + ks * 32, LDQ, lane);
      // B(k=depth, n=key) = Kchunk[key][depth] == transposed [n][k] storage
      v16bf bf = frag_b_bf16(kv + (wn * 16) * LDK + ks * 32, LDK, lane);
      acc = WMMA_BF16(a, bf, acc);
    }
    const int gcol = ch * 64 + wn * 16 + (lane & 15);
    const float mv = mask[(size_t)b * Sc + gcol] * -1e9f;
#pragma unroll
    for (int r = 0; r < 8; ++r) {
      int row = wm * 16 + r + 8 * (lane >> 4);
      sc[row * 1024 + gcol] = acc[r] * 0.125f + mv;  // 1/sqrt(64)
    }
  }
  __syncthreads();

  // ---- Softmax over 1024 cols: 8 threads per row, 128 cols each ----
  {
    const int r = tid >> 3, p = tid & 7;
    float* rowp = sc + r * 1024 + p * 128;
    float mx = -3.4e38f;
#pragma unroll 8
    for (int c = 0; c < 128; ++c) mx = fmaxf(mx, rowp[c]);
    red[r * 8 + p] = mx;
    __syncthreads();
    if (p == 0) {
      float m2 = red[r * 8];
#pragma unroll
      for (int j = 1; j < 8; ++j) m2 = fmaxf(m2, red[r * 8 + j]);
      rowmax[r] = m2;
    }
    __syncthreads();
    const float rm = rowmax[r];
    float sum = 0.f;
#pragma unroll 8
    for (int c = 0; c < 128; ++c) {
      float e = __expf(rowp[c] - rm);
      rowp[c] = e;
      sum += e;
    }
    red[r * 8 + p] = sum;
    __syncthreads();
    if (p == 0) {
      float s2 = 0.f;
#pragma unroll
      for (int j = 0; j < 8; ++j) s2 += red[r * 8 + j];
      rowsum[r] = s2;
    }
    __syncthreads();
    const float inv = 1.0f / rowsum[r];
    float* gattn = attn_out + ((((size_t)b * Hc) + h) * Sc + q0 + r) * Sc + p * 128;
    __bf16* pb = pbuf + r * LDP + p * 128;
#pragma unroll 4
    for (int c = 0; c < 128; c += 2) {
      float e0 = rowp[c] * inv, e1 = rowp[c + 1] * inv;
      float2 o; o.x = e0; o.y = e1;
      *(float2*)(gattn + c) = o;                 // materialize attn exactly once
      *(unsigned*)(pb + c) = pack_bf16(e0, e1);  // bf16 probs for pass 2
    }
  }

  // ---- Pass 2: out(32x64) = attn(32x1024) @ V(1024x64) ----
  v8f oacc = {};
  for (int ch = 0; ch < 16; ++ch) {
    __syncthreads();
    // Stage V chunk transposed: V^T [depth][key] (manual: transpose needed)
#pragma unroll
    for (int i = 0; i < 16; ++i) {
      int e = tid + i * 256;
      int key = e >> 6, dep = e & 63;
      kv[dep * LDK + key] = Vb[(size_t)(ch * 64 + key) * DEPTH + dep];
    }
    if (ch + 1 < 16)
      __builtin_prefetch(Vb + (size_t)(ch * 64 + 64 + (tid >> 2)) * DEPTH + (tid & 3) * 16, 0, 0);
    __syncthreads();
#pragma unroll
    for (int ks = 0; ks < 2; ++ks) {
      v16bf a = frag_a_bf16(pbuf + (wm * 16) * LDP + ch * 64 + ks * 32, LDP, lane);
      // B(k=key, n=depth) = V[key][depth] == V^T[n][k] storage
      v16bf bf = frag_b_bf16(kv + (wn * 16) * LDK + ks * 32, LDK, lane);
      oacc = WMMA_BF16(a, bf, oacc);
    }
  }
  const int dcol = h * DEPTH + wn * 16 + (lane & 15);
#pragma unroll
  for (int r = 0; r < 8; ++r) {
    int row = q0 + wm * 16 + r + 8 * (lane >> 4);
    concat[((size_t)b * Sc + row) * Dc + dcol] = (__bf16)oacc[r];
  }
}

// ---------------------------------------------------------------------------
// Host launch
// ---------------------------------------------------------------------------
extern "C" void kernel_launch(void* const* d_in, const int* in_sizes, int n_in,
                              void* d_out, int out_size, void* d_ws, size_t ws_size,
                              hipStream_t stream) {
  const float* v    = (const float*)d_in[0];
  const float* k    = (const float*)d_in[1];
  const float* q    = (const float*)d_in[2];
  const float* mask = (const float*)d_in[3];
  const float* Wq = (const float*)d_in[4];  const float* bq = (const float*)d_in[5];
  const float* Wk = (const float*)d_in[6];  const float* bk = (const float*)d_in[7];
  const float* Wv = (const float*)d_in[8];  const float* bv = (const float*)d_in[9];
  const float* Wo = (const float*)d_in[10]; const float* bo = (const float*)d_in[11];

  float* out  = (float*)d_out;                       // [B,S,D] f32
  float* attn = out + (size_t)Bc * Sc * Dc;          // [B,H,S,S] f32

  const size_t nBHSD = (size_t)Bc * Hc * Sc * DEPTH; // 8,388,608 elements
  __bf16* Qh     = (__bf16*)d_ws;
  __bf16* Kh     = Qh + nBHSD;
  __bf16* Vh     = Kh + nBHSD;
  __bf16* concat = Vh + nBHSD;                       // [B,S,D] bf16

  const dim3 blk(256);
  const dim3 ggrid(Dc / 128, (Bc * Sc) / 128);       // 8 x 64 blocks

  // QKV projections (f32 in, head-split bf16 out)
  wmma_gemm_kernel<false, true><<<ggrid, blk, 0, stream>>>(q, Wq, bq, Qh);
  wmma_gemm_kernel<false, true><<<ggrid, blk, 0, stream>>>(k, Wk, bk, Kh);
  wmma_gemm_kernel<false, true><<<ggrid, blk, 0, stream>>>(v, Wv, bv, Vh);

  // Fused scores + softmax + attn@V
  attention_kernel<<<dim3(Sc / 32, Hc, Bc), blk, ATTN_SMEM, stream>>>(
      Qh, Kh, Vh, mask, attn, concat);

  // Output projection (bf16 in, f32 out)
  wmma_gemm_kernel<true, false><<<ggrid, blk, 0, stream>>>(concat, Wo, bo, out);

  (void)in_sizes; (void)n_in; (void)out_size; (void)ws_size;
}
